// TransformerModel_29231547416883
// MI455X (gfx1250) — compile-verified, matrix operand order checked by reading, not run on hardware
//
#include <hip/hip_runtime.h>

// ---------------------------------------------------------------------------
// CDNA5 (gfx1250) implementation: all GEMMs via v_wmma_f32_16x16x32_bf16.
// Each wave computes a 16x64 strip of C (4 N-tiles), reusing the A operand
// registers across 4 back-to-back WMMAs (2.5 loads/WMMA vs 4).
// Padded shapes: 300->320(K)/320(N), 900->960, 100->128, 200->256.
// ---------------------------------------------------------------------------

typedef __bf16 bf16_t;
typedef __attribute__((ext_vector_type(16))) __bf16 bf16x16;
typedef __attribute__((ext_vector_type(8)))  __bf16 bf16x8;
typedef __attribute__((ext_vector_type(8)))  float  f32x8;

constexpr int SEQ  = 4096;
constexpr int EMB  = 300;
constexpr int EMBP = 320;   // padded embed (K mult of 32, N mult of 64)
constexpr int QKVP = 960;   // 3*300 padded to mult of 64
constexpr int HDV  = 100;   // head dim
constexpr int HDP  = 128;   // padded head dim
constexpr int HIDV = 200;   // FF hidden
constexpr int HIDP = 256;   // padded FF hidden (N and K mult of 64/32)

// ---------------------------------------------------------------------------
// WMMA GEMM:  C[M,N] = scale * (A[M,K] * B[N,K]^T) + bias  (+causal/relu)
// A, B bf16 row-major, zero-padded; K mult of 32; N mult of 64; M mult of 64.
// One wave -> 16x64 of C. blockDim (32,4): 4 waves stack along M.
// ---------------------------------------------------------------------------
__global__ void gemm_bf16_wmma(const bf16_t* __restrict__ A, int lda,
                               const bf16_t* __restrict__ B, int ldb,
                               float* __restrict__ C, int ldc,
                               bf16_t* __restrict__ Cb, int ldcb,
                               const float* __restrict__ bias, int bias_n,
                               int Kp, float scale, int causal, int do_relu,
                               int col_off, int n_store)
{
  const int lane = threadIdx.x;       // 0..31 (wave32)
  const int half = lane >> 4;         // 0/1
  const int r    = lane & 15;
  const int m0 = (blockIdx.y * 4 + threadIdx.y) * 16;
  const int n0 = blockIdx.x * 64;

  const bf16_t* Arow = A + (size_t)(m0 + r) * lda + half * 8;
  const bf16_t* Brow = B + (size_t)(n0 + r) * ldb + half * 8;
  const size_t bstep = (size_t)16 * ldb;          // 16 rows of B per N-tile

  f32x8 acc0 = {0.f,0.f,0.f,0.f,0.f,0.f,0.f,0.f};
  f32x8 acc1 = acc0, acc2 = acc0, acc3 = acc0;

  for (int k = 0; k < Kp; k += 32) {
    __builtin_prefetch(Arow + k + 128, 0, 3);     // WGP-scope prefetch intent
    __builtin_prefetch(Brow + k + 128, 0, 3);
    bf16x8 alo = *(const bf16x8*)(Arow + k);
    bf16x8 ahi = *(const bf16x8*)(Arow + k + 16);
    bf16x16 av = __builtin_shufflevector(alo, ahi, 0,1,2,3,4,5,6,7,8,9,10,11,12,13,14,15);
    #pragma unroll
    for (int j = 0; j < 4; ++j) {
      const bf16_t* bp = Brow + (size_t)j * bstep + k;
      bf16x8 blo = *(const bf16x8*)(bp);
      bf16x8 bhi = *(const bf16x8*)(bp + 16);
      bf16x16 bv = __builtin_shufflevector(blo, bhi, 0,1,2,3,4,5,6,7,8,9,10,11,12,13,14,15);
      f32x8 cin = (j == 0) ? acc0 : (j == 1) ? acc1 : (j == 2) ? acc2 : acc3;
      f32x8 cout = __builtin_amdgcn_wmma_f32_16x16x32_bf16(false, av, false, bv,
                                                           (short)0, cin, false, false);
      if (j == 0) acc0 = cout; else if (j == 1) acc1 = cout;
      else if (j == 2) acc2 = cout; else acc3 = cout;
    }
  }

  // epilogue: element i of tile j -> M = m0 + half*8 + i, N = n0 + j*16 + r
  #pragma unroll
  for (int j = 0; j < 4; ++j) {
    const int n = n0 + j * 16 + r;
    if (n >= n_store) continue;
    const float bval = (bias && n < bias_n) ? bias[n] : 0.f;
    const f32x8 a = (j == 0) ? acc0 : (j == 1) ? acc1 : (j == 2) ? acc2 : acc3;
    #pragma unroll
    for (int i = 0; i < 8; ++i) {
      int m = m0 + half * 8 + i;
      float v = a[i] * scale + bval;
      if (causal && n > m) v = -1e30f;
      if (do_relu) v = fmaxf(v, 0.f);
      if (C)  C[(size_t)m * ldc  + col_off + n] = v;
      if (Cb) Cb[(size_t)m * ldcb + col_off + n] = (bf16_t)v;
    }
  }
}

// x = src + positional encoding; write padded f32 + bf16 copies.
__global__ void add_pe(const float* __restrict__ src,
                       float* __restrict__ x, bf16_t* __restrict__ xb)
{
  const int s = blockIdx.x, c = threadIdx.x;       // blockDim = EMBP
  float v = 0.f;
  if (c < EMB) {
    float dv  = __expf(-(float)(c & ~1) * (9.210340371976184f / (float)EMB));
    float ang = (float)s * dv;
    v = src[(size_t)s * EMB + c] + ((c & 1) ? __cosf(ang) : __sinf(ang));
  }
  x [(size_t)s * EMBP + c] = v;
  xb[(size_t)s * EMBP + c] = (bf16_t)v;
}

// dst[r,c<Cp] = (r<R && c<Cv) ? bf16(src[r*lds+c]) : 0    (grid.y = Rp)
__global__ void pad_cast_bf16(const float* __restrict__ src, int lds, int R, int Cv,
                              bf16_t* __restrict__ dst, int Cp)
{
  const int c = blockIdx.x * blockDim.x + threadIdx.x;
  const int r = blockIdx.y;
  if (c >= Cp) return;
  float v = (r < R && c < Cv) ? src[(size_t)r * lds + c] : 0.f;
  dst[(size_t)r * Cp + c] = (bf16_t)v;
}

// Slice one attention head out of qkv; V is written transposed [HDP, SEQ].
__global__ void extract_head(const float* __restrict__ qkv, int h,
                             bf16_t* __restrict__ qh, bf16_t* __restrict__ kh,
                             bf16_t* __restrict__ vhT)
{
  const int s = blockIdx.x, j = threadIdx.x;       // blockDim = HDP
  float q = 0.f, k = 0.f, v = 0.f;
  if (j < HDV) {
    const float* row = qkv + (size_t)s * QKVP;
    q = row[h * HDV + j];
    k = row[EMB     + h * HDV + j];
    v = row[2 * EMB + h * HDV + j];
  }
  qh [(size_t)s * HDP + j] = (bf16_t)q;
  kh [(size_t)s * HDP + j] = (bf16_t)k;
  vhT[(size_t)j * SEQ + s] = (bf16_t)v;
}

// Row softmax (f32 in, bf16 out). One block (256 thr) per row.
__global__ void softmax_row_bf16(const float* __restrict__ X,
                                 bf16_t* __restrict__ O, int n)
{
  const int row = blockIdx.x, t = threadIdx.x;
  const float* x = X + (size_t)row * n;
  __shared__ float red[256];
  float m = -3.0e38f;
  for (int j = t; j < n; j += 256) m = fmaxf(m, x[j]);
  red[t] = m; __syncthreads();
  for (int st = 128; st > 0; st >>= 1) { if (t < st) red[t] = fmaxf(red[t], red[t + st]); __syncthreads(); }
  m = red[0]; __syncthreads();
  float s = 0.f;
  for (int j = t; j < n; j += 256) s += __expf(x[j] - m);
  red[t] = s; __syncthreads();
  for (int st = 128; st > 0; st >>= 1) { if (t < st) red[t] += red[t + st]; __syncthreads(); }
  const float inv = 1.f / red[0];
  bf16_t* o = O + (size_t)row * n;
  for (int j = t; j < n; j += 256) o[j] = (bf16_t)(__expf(x[j] - m) * inv);
}

// out[C,R] = in[R,C]^T  (bf16, LDS-tiled). grid = (ceil(C/32), ceil(R/32)), block (32,8)
__global__ void transpose_bf16(const bf16_t* __restrict__ in, bf16_t* __restrict__ out,
                               int R, int C)
{
  __shared__ bf16_t t[32][33];
  const int c0 = blockIdx.x * 32, r0 = blockIdx.y * 32;
  const int tx = threadIdx.x, ty = threadIdx.y;
  for (int i = 0; i < 32; i += 8) {
    int r = r0 + ty + i, c = c0 + tx;
    t[ty + i][tx] = (r < R && c < C) ? in[(size_t)r * C + c] : (bf16_t)0.f;
  }
  __syncthreads();
  for (int i = 0; i < 32; i += 8) {
    int r = c0 + ty + i, c = r0 + tx;
    if (r < C && c < R) out[(size_t)r * R + c] = t[tx][ty + i];
  }
}

// y = LayerNorm(xa + xr) * g + b  (one block of 512 per row; 300 valid cols)
__global__ void add_ln(const float* __restrict__ xa, const float* __restrict__ xr,
                       const float* __restrict__ g, const float* __restrict__ b,
                       float* __restrict__ y, bf16_t* __restrict__ yb)
{
  const int s = blockIdx.x, c = threadIdx.x;       // blockDim = 512
  __shared__ float red[512];
  float v = 0.f;
  if (c < EMB) v = xa[(size_t)s * EMBP + c] + xr[(size_t)s * EMBP + c];
  red[c] = v; __syncthreads();
  for (int st = 256; st > 0; st >>= 1) { if (c < st) red[c] += red[c + st]; __syncthreads(); }
  const float mu = red[0] * (1.f / (float)EMB); __syncthreads();
  const float d = (c < EMB) ? v - mu : 0.f;
  red[c] = d * d; __syncthreads();
  for (int st = 256; st > 0; st >>= 1) { if (c < st) red[c] += red[c + st]; __syncthreads(); }
  const float inv = rsqrtf(red[0] * (1.f / (float)EMB) + 1e-5f);
  if (c < EMBP) {
    float o = (c < EMB) ? d * inv * g[c] + b[c] : 0.f;
    y [(size_t)s * EMBP + c] = o;
    yb[(size_t)s * EMBP + c] = (bf16_t)o;
  }
}

// inp[c] = sum_s max(X[s,c'], Y[s,c'])  where c<300 -> (H,P), else (G,Q).
__global__ void colmax_sum(const float* __restrict__ H, const float* __restrict__ P,
                           const float* __restrict__ G, const float* __restrict__ Q,
                           float* __restrict__ inp)
{
  const int c = blockIdx.x;                        // 0..599
  const float* X = (c < EMB) ? H : G;
  const float* Y = (c < EMB) ? P : Q;
  const int cc = (c < EMB) ? c : c - EMB;
  __shared__ float red[256];
  float acc = 0.f;
  for (int s = threadIdx.x; s < SEQ; s += 256)
    acc += fmaxf(X[(size_t)s * EMBP + cc], Y[(size_t)s * EMBP + cc]);
  red[threadIdx.x] = acc; __syncthreads();
  for (int st = 128; st > 0; st >>= 1) { if (threadIdx.x < st) red[threadIdx.x] += red[threadIdx.x + st]; __syncthreads(); }
  if (threadIdx.x == 0) inp[c] = red[0];
}

// out[0] = fc2( relu( fc1 @ inp + b1 ) ) + b2   — single block of 512.
__global__ void fc_head(const float* __restrict__ inp,
                        const float* __restrict__ w1, const float* __restrict__ b1,
                        const float* __restrict__ w2, const float* __restrict__ b2,
                        float* __restrict__ out)
{
  __shared__ float si[600];
  __shared__ float h1[300];
  __shared__ float red[512];
  const int t = threadIdx.x;
  for (int j = t; j < 600; j += 512) si[j] = inp[j];
  __syncthreads();
  if (t < 300) {
    float a = b1[t];
    for (int j = 0; j < 600; ++j) a += w1[t * 600 + j] * si[j];
    h1[t] = fmaxf(a, 0.f);
  }
  __syncthreads();
  red[t] = (t < 300) ? w2[t] * h1[t] : 0.f; __syncthreads();
  for (int st = 256; st > 0; st >>= 1) { if (t < st) red[t] += red[t + st]; __syncthreads(); }
  if (t == 0) out[0] = red[0] + b2[0];
}

// ---------------------------------------------------------------------------
extern "C" void kernel_launch(void* const* d_in, const int* in_sizes, int n_in,
                              void* d_out, int out_size, void* d_ws, size_t ws_size,
                              hipStream_t stream)
{
  const float* solv = (const float*)d_in[0];
  const float* solu = (const float*)d_in[1];
  const float* ipw  = (const float*)d_in[2];   // [2,900,300]
  const float* ipb  = (const float*)d_in[3];   // [2,900]
  const float* oww  = (const float*)d_in[4];   // [2,300,300]
  const float* owb  = (const float*)d_in[5];   // [2,300]
  const float* l1g  = (const float*)d_in[6];
  const float* l1b  = (const float*)d_in[7];
  const float* w1   = (const float*)d_in[8];   // [2,200,300]
  const float* b1   = (const float*)d_in[9];   // [2,200]
  const float* w2   = (const float*)d_in[10];  // [2,300,200]
  const float* b2   = (const float*)d_in[11];  // [2,300]
  const float* l2g  = (const float*)d_in[12];
  const float* l2b  = (const float*)d_in[13];
  const float* fc1w = (const float*)d_in[14];
  const float* fc1b = (const float*)d_in[15];
  const float* fc2w = (const float*)d_in[16];
  const float* fc2b = (const float*)d_in[17];
  (void)in_sizes; (void)n_in; (void)out_size; (void)ws_size;

  char* base = (char*)d_ws;
  size_t off = 0;
  auto carve = [&](size_t bytes) -> char* {
    char* p = base + off;
    off += (bytes + 255) & ~(size_t)255;
    return p;
  };

  float*  x0[2]  = {(float*)carve((size_t)SEQ*EMBP*4), (float*)carve((size_t)SEQ*EMBP*4)};
  float*  xf[2]  = {(float*)carve((size_t)SEQ*EMBP*4), (float*)carve((size_t)SEQ*EMBP*4)};
  bf16_t* xb     = (bf16_t*)carve((size_t)SEQ*EMBP*2);
  bf16_t* hgb[2] = {(bf16_t*)carve((size_t)SEQ*EMBP*2), (bf16_t*)carve((size_t)SEQ*EMBP*2)};
  bf16_t* hgt[2] = {(bf16_t*)carve((size_t)EMBP*SEQ*2), (bf16_t*)carve((size_t)EMBP*SEQ*2)};
  float*  qkv    = (float*)carve((size_t)SEQ*QKVP*4);
  float*  obuf   = (float*)carve((size_t)SEQ*EMBP*4);
  bf16_t* ob     = (bf16_t*)carve((size_t)SEQ*EMBP*2);
  float*  attno  = (float*)carve((size_t)SEQ*EMBP*4);
  float*  x1     = (float*)carve((size_t)SEQ*EMBP*4);
  bf16_t* x1b    = (bf16_t*)carve((size_t)SEQ*EMBP*2);
  bf16_t* f1b    = (bf16_t*)carve((size_t)SEQ*HIDP*2);
  float*  f2     = (float*)carve((size_t)SEQ*EMBP*4);
  bf16_t* qh     = (bf16_t*)carve((size_t)SEQ*HDP*2);
  bf16_t* kh     = (bf16_t*)carve((size_t)SEQ*HDP*2);
  bf16_t* vhT    = (bf16_t*)carve((size_t)HDP*SEQ*2);
  float*  Pm     = (float*)carve((size_t)SEQ*EMBP*4);
  float*  Qm     = (float*)carve((size_t)SEQ*EMBP*4);
  bf16_t* wqkvb  = (bf16_t*)carve((size_t)QKVP*EMBP*2);
  bf16_t* woutb  = (bf16_t*)carve((size_t)EMBP*EMBP*2);
  bf16_t* wff1b  = (bf16_t*)carve((size_t)HIDP*EMBP*2);
  bf16_t* wff2b  = (bf16_t*)carve((size_t)EMBP*HIDP*2);
  float*  inp    = (float*)carve(600*4);
  float*  sc     = (float*)carve((size_t)SEQ*SEQ*4);   // scores (f32)
  bf16_t* attnb  = (bf16_t*)carve((size_t)SEQ*SEQ*2);  // softmax out (bf16)
  bf16_t* aT     = (bf16_t*)sc;                        // alias: a^T over dead scores

  auto gemm = [&](const bf16_t* A, int lda, const bf16_t* B, int ldb,
                  float* C, int ldc, bf16_t* Cb, int ldcb,
                  const float* bias, int bias_n, int Np, int Kp,
                  float scale, int causal, int relu, int col_off, int n_store) {
    dim3 blk(32, 4), grd(Np / 64, SEQ / 64);
    gemm_bf16_wmma<<<grd, blk, 0, stream>>>(A, lda, B, ldb, C, ldc, Cb, ldcb,
                                            bias, bias_n, Kp, scale, causal, relu,
                                            col_off, n_store);
  };

  for (int e = 0; e < 2; ++e) {
    // weight padding / casting to bf16
    pad_cast_bf16<<<dim3((EMBP+127)/128, QKVP), 128, 0, stream>>>(ipw + (size_t)e*3*EMB*EMB, EMB, 3*EMB, EMB, wqkvb, EMBP);
    pad_cast_bf16<<<dim3((EMBP+127)/128, EMBP), 128, 0, stream>>>(oww + (size_t)e*EMB*EMB, EMB, EMB, EMB, woutb, EMBP);
    pad_cast_bf16<<<dim3((EMBP+127)/128, HIDP), 128, 0, stream>>>(w1 + (size_t)e*HIDV*EMB, EMB, HIDV, EMB, wff1b, EMBP);
    pad_cast_bf16<<<dim3((HIDP+127)/128, EMBP), 128, 0, stream>>>(w2 + (size_t)e*EMB*HIDV, HIDV, EMB, HIDV, wff2b, HIDP);

    const float* src = e ? solu : solv;
    add_pe<<<SEQ, EMBP, 0, stream>>>(src, x0[e], xb);

    // qkv = x @ Wqkv^T + b
    gemm(xb, EMBP, wqkvb, EMBP, qkv, QKVP, nullptr, 0,
         ipb + (size_t)e*3*EMB, 3*EMB, QKVP, EMBP, 1.f, 0, 0, 0, QKVP);

    for (int h = 0; h < 3; ++h) {
      extract_head<<<SEQ, HDP, 0, stream>>>(qkv, h, qh, kh, vhT);
      // scores = Q K^T / 10 + causal mask
      gemm(qh, HDP, kh, HDP, sc, SEQ, nullptr, 0,
           nullptr, 0, SEQ, HDP, 0.1f, 1, 0, 0, SEQ);
      softmax_row_bf16<<<SEQ, 256, 0, stream>>>(sc, attnb, SEQ);
      // o[:, h*100 : h*100+100] = attn @ V
      gemm(attnb, SEQ, vhT, SEQ, obuf, EMBP, nullptr, 0,
           nullptr, 0, HDP, SEQ, 1.f, 0, 0, h * HDV, HDV);
    }

    pad_cast_bf16<<<dim3((EMBP+127)/128, SEQ), 128, 0, stream>>>(obuf, EMBP, SEQ, EMB, ob, EMBP);
    // attn_out = o @ Wout^T + b
    gemm(ob, EMBP, woutb, EMBP, attno, EMBP, nullptr, 0,
         owb + (size_t)e*EMB, EMB, EMBP, EMBP, 1.f, 0, 0, 0, EMBP);
    add_ln<<<SEQ, 512, 0, stream>>>(x0[e], attno, l1g + (size_t)e*EMB, l1b + (size_t)e*EMB, x1, x1b);

    // ff = relu(x1 @ W1^T + b1) @ W2^T + b2
    gemm(x1b, EMBP, wff1b, EMBP, nullptr, 0, f1b, HIDP,
         b1 + (size_t)e*HIDV, HIDV, HIDP, EMBP, 1.f, 0, 1, 0, HIDP);
    gemm(f1b, HIDP, wff2b, HIDP, f2, EMBP, nullptr, 0,
         b2 + (size_t)e*EMB, EMB, EMBP, HIDP, 1.f, 0, 0, 0, EMBP);
    add_ln<<<SEQ, 512, 0, stream>>>(x1, f2, l2g + (size_t)e*EMB, l2b + (size_t)e*EMB, xf[e], hgb[e]);
  }

  // ---- co-attention ----
  gemm(hgb[0], EMBP, hgb[1], EMBP, sc, SEQ, nullptr, 0,
       nullptr, 0, SEQ, EMBP, 1.f, 0, 0, 0, SEQ);            // a_score = H G^T
  softmax_row_bf16<<<SEQ, 256, 0, stream>>>(sc, attnb, SEQ); // a (bf16)
  transpose_bf16<<<dim3(SEQ/32, SEQ/32), dim3(32, 8), 0, stream>>>(attnb, aT, SEQ, SEQ);
  transpose_bf16<<<dim3(EMBP/32, SEQ/32), dim3(32, 8), 0, stream>>>(hgb[0], hgt[0], SEQ, EMBP);
  transpose_bf16<<<dim3(EMBP/32, SEQ/32), dim3(32, 8), 0, stream>>>(hgb[1], hgt[1], SEQ, EMBP);
  gemm(attnb, SEQ, hgt[1], SEQ, Pm, EMBP, nullptr, 0,
       nullptr, 0, EMBP, SEQ, 1.f, 0, 0, 0, EMBP);           // P = a @ G
  gemm(aT, SEQ, hgt[0], SEQ, Qm, EMBP, nullptr, 0,
       nullptr, 0, EMBP, SEQ, 1.f, 0, 0, 0, EMBP);           // Q = a^T @ H

  colmax_sum<<<600, 256, 0, stream>>>(xf[0], Pm, xf[1], Qm, inp);
  fc_head<<<1, 512, 0, stream>>>(inp, fc1w, fc1b, fc2w, fc2b, (float*)d_out);
}
